// CenterHeadIoUDebug_9019431322219
// MI455X (gfx1250) — matrix operand order repaired and checked by reading, not compiled, not run on hardware
//
#include <hip/hip_runtime.h>
#include <hip/hip_bf16.h>

typedef __attribute__((ext_vector_type(16))) _Float16 v16h;
typedef __attribute__((ext_vector_type(8)))  _Float16 v8h;
typedef __attribute__((ext_vector_type(8)))  float    v8f;

#define NB   4
#define CIN  512
#define HH   188
#define WW   188
#define SCH  64
#define NCLS 3
#define NM   200
#define NMAX 500

// ---------------- weight packing ----------------
// w_shared [64][512][3][3] f32 -> pws [9 tap][16 cc][64 oc][32 cw] f16
__global__ __launch_bounds__(256) void pack_shared(const float* __restrict__ w,
                                                   _Float16* __restrict__ o) {
  int t = blockIdx.x * 256 + threadIdx.x;
  if (t >= 9 * 16 * 64 * 32) return;
  int cw = t & 31;
  int oc = (t >> 5) & 63;
  int cc = (t >> 11) & 15;
  int tap = t >> 15;
  int ky = tap / 3, kx = tap - ky * 3;
  int ci = cc * 32 + cw;
  o[t] = (_Float16)w[((oc * CIN + ci) * 3 + ky) * 3 + kx];
}

// w1_heads [6][64][64][3][3] -> pw1 [6][9][2][64][32]
__global__ __launch_bounds__(256) void pack_head1(const float* __restrict__ w,
                                                  _Float16* __restrict__ o) {
  int t = blockIdx.x * 256 + threadIdx.x;
  if (t >= 6 * 9 * 2 * 64 * 32) return;
  int cw = t & 31;
  int oc = (t >> 5) & 63;
  int cc = (t >> 11) & 1;
  int tap = (t >> 12) % 9;
  int head = t / 36864;
  int ky = tap / 3, kx = tap - ky * 3;
  int ci = cc * 32 + cw;
  o[t] = (_Float16)w[(((head * 64 + oc) * 64 + ci) * 3 + ky) * 3 + kx];
}

// w2_heads [6][3][64][3][3] -> pw2 [6][9][2][16 oc][32], oc>=3 zero-padded
__global__ __launch_bounds__(256) void pack_head2(const float* __restrict__ w,
                                                  _Float16* __restrict__ o) {
  int t = blockIdx.x * 256 + threadIdx.x;
  if (t >= 6 * 9 * 2 * 16 * 32) return;
  int cw = t & 31;
  int oc = (t >> 5) & 15;
  int cc = (t >> 9) & 1;
  int tap = (t >> 10) % 9;
  int head = t / 9216;
  int ky = tap / 3, kx = tap - ky * 3;
  int ci = cc * 32 + cw;
  float v = 0.f;
  if (oc < 3) v = w[(((head * 3 + oc) * 64 + ci) * 3 + ky) * 3 + kx];
  o[t] = (_Float16)v;
}

// ---- async staging of f16 NHWC halo tile [3][66][32ch] into LDS ----
__device__ __forceinline__ void stage_tile_f16_async(const _Float16* __restrict__ src,
                                                     _Float16* smem, int b, int y,
                                                     int x0, int cc, int tid) {
  // 3*66*4 chunks of 8 halves (16B) each
  for (int j = tid; j < 3 * 66 * 4; j += 128) {
    int c8 = j & 3;
    int xx = (j >> 2) % 66;
    int row = j / (66 * 4);
    int gy = y + row - 1;
    int gx = x0 + xx - 1;
    _Float16* lp = &smem[(row * 66 + xx) * 32 + c8 * 8];
    if (gy >= 0 && gy < HH && gx >= 0 && gx < WW) {
      const _Float16* gp = src + (((size_t)b * HH + gy) * WW + gx) * 64 + cc * 32 + c8 * 8;
      unsigned lds_off = (unsigned)(size_t)lp;
      asm volatile("global_load_async_to_lds_b128 %0, %1, off"
                   :: "v"(lds_off), "v"((unsigned long long)(size_t)gp)
                   : "memory");
    } else {
      v8h z = {};
      *(v8h*)lp = z;
    }
  }
  asm volatile("s_wait_asynccnt 0" ::: "memory");
}

// ---------------- shared conv: f32 NCHW -> (conv+BN+ReLU) -> f16 NHWC ----------------
// workgroup: 128 thr = 4 waves; tile 64 positions x 64 oc; wave = one 16-oc tile, all 64 pos
__global__ __launch_bounds__(128) void conv_shared(const float* __restrict__ src,
                                                   const _Float16* __restrict__ wpk,
                                                   const float* __restrict__ bn,
                                                   _Float16* __restrict__ dst) {
  __shared__ _Float16 smem[3 * 66 * 32];
  const int x0 = blockIdx.x * 64;
  const int y  = blockIdx.y;
  const int b  = blockIdx.z;
  const int tid = threadIdx.x;
  const int wave = tid >> 5;
  const int lane = tid & 31;
  const int lhalf = lane >> 4;  // 0/1
  const int lidx  = lane & 15;

  v8f acc[4] = {};
  for (int cc = 0; cc < 16; ++cc) {
    __syncthreads();
    for (int i = tid; i < 3 * 66 * 32; i += 128) {
      int xx = i % 66;
      int rest = i / 66;
      int cw = rest & 31;
      int row = rest >> 5;
      int gy = y + row - 1;
      int gx = x0 + xx - 1;
      float v = 0.f;
      if (gy >= 0 && gy < HH && gx >= 0 && gx < WW)
        v = src[((b * CIN + (cc * 32 + cw)) * HH + gy) * WW + gx];
      smem[(row * 66 + xx) * 32 + cw] = (_Float16)v;
    }
    __syncthreads();
#pragma unroll
    for (int tap = 0; tap < 9; ++tap) {
      int ky = tap / 3, kx = tap - ky * 3;
      // one B fragment per wave per tap (16 oc), reused by 4 WMMAs
      const _Float16* bp = wpk + ((size_t)(tap * 16 + cc) * 64 + wave * 16 + lidx) * 32 + lhalf * 16;
      v16h Bf = *(const v16h*)bp;
      union { v16h v; v8h h[2]; } A[4];
#pragma unroll
      for (int p = 0; p < 4; ++p) {
        const _Float16* ap = &smem[(ky * 66 + (p * 16 + lidx + kx)) * 32 + lhalf * 16];
        A[p].h[0] = *(const v8h*)ap;
        A[p].h[1] = *(const v8h*)(ap + 8);
      }
#pragma unroll
      for (int p = 0; p < 4; ++p)
        acc[p] = __builtin_amdgcn_wmma_f32_16x16x32_f16(false, A[p].v, false, Bf,
                                                        (short)0, acc[p], false, false);
    }
  }
  int oc = wave * 16 + lidx;
  float scale = bn[oc] * rsqrtf(bn[192 + oc] + 1e-5f);
  float beta = bn[64 + oc], mu = bn[128 + oc];
#pragma unroll
  for (int p = 0; p < 4; ++p) {
#pragma unroll
    for (int r = 0; r < 8; ++r) {
      int gx = x0 + p * 16 + r + lhalf * 8;
      if (gx < WW) {
        float v = (acc[p][r] - mu) * scale + beta;
        v = fmaxf(v, 0.f);
        dst[(((size_t)b * HH + y) * WW + gx) * 64 + oc] = (_Float16)v;
      }
    }
  }
}

// ---------------- head conv1: f16 NHWC 64 -> (conv+BN+ReLU) -> f16 NHWC 64 ----------------
__global__ __launch_bounds__(128) void conv_head1(const _Float16* __restrict__ src,
                                                  const _Float16* __restrict__ wpk,
                                                  const float* __restrict__ bn,
                                                  _Float16* __restrict__ dst) {
  __shared__ _Float16 smem[3 * 66 * 32];
  const int x0 = blockIdx.x * 64;
  const int y  = blockIdx.y;
  const int b  = blockIdx.z;
  const int tid = threadIdx.x;
  const int wave = tid >> 5;
  const int lane = tid & 31;
  const int lhalf = lane >> 4;
  const int lidx  = lane & 15;

  v8f acc[4] = {};
  for (int cc = 0; cc < 2; ++cc) {
    __syncthreads();
    stage_tile_f16_async(src, smem, b, y, x0, cc, tid);
    __syncthreads();
#pragma unroll
    for (int tap = 0; tap < 9; ++tap) {
      int ky = tap / 3, kx = tap - ky * 3;
      const _Float16* bp = wpk + ((size_t)(tap * 2 + cc) * 64 + wave * 16 + lidx) * 32 + lhalf * 16;
      v16h Bf = *(const v16h*)bp;
      union { v16h v; v8h h[2]; } A[4];
#pragma unroll
      for (int p = 0; p < 4; ++p) {
        const _Float16* ap = &smem[(ky * 66 + (p * 16 + lidx + kx)) * 32 + lhalf * 16];
        A[p].h[0] = *(const v8h*)ap;
        A[p].h[1] = *(const v8h*)(ap + 8);
      }
#pragma unroll
      for (int p = 0; p < 4; ++p)
        acc[p] = __builtin_amdgcn_wmma_f32_16x16x32_f16(false, A[p].v, false, Bf,
                                                        (short)0, acc[p], false, false);
    }
  }
  int oc = wave * 16 + lidx;
  float scale = bn[oc] * rsqrtf(bn[192 + oc] + 1e-5f);
  float beta = bn[64 + oc], mu = bn[128 + oc];
#pragma unroll
  for (int p = 0; p < 4; ++p) {
#pragma unroll
    for (int r = 0; r < 8; ++r) {
      int gx = x0 + p * 16 + r + lhalf * 8;
      if (gx < WW) {
        float v = (acc[p][r] - mu) * scale + beta;
        v = fmaxf(v, 0.f);
        dst[(((size_t)b * HH + y) * WW + gx) * 64 + oc] = (_Float16)v;
      }
    }
  }
}

// ---------------- head conv2: f16 NHWC 64 -> conv(16 oc) + bias -> head_out channels ----------------
__global__ __launch_bounds__(128) void conv_head2(const _Float16* __restrict__ src,
                                                  const _Float16* __restrict__ wpk,
                                                  const float* __restrict__ bias,
                                                  float* __restrict__ out,
                                                  int choff, int ccnt) {
  __shared__ _Float16 smem[3 * 66 * 32];
  const int x0 = blockIdx.x * 64;
  const int y  = blockIdx.y;
  const int b  = blockIdx.z;
  const int tid = threadIdx.x;
  const int wave = tid >> 5;
  const int lane = tid & 31;
  const int lhalf = lane >> 4;
  const int lidx  = lane & 15;

  v8f acc = {};
  for (int cc = 0; cc < 2; ++cc) {
    __syncthreads();
    stage_tile_f16_async(src, smem, b, y, x0, cc, tid);
    __syncthreads();
#pragma unroll
    for (int tap = 0; tap < 9; ++tap) {
      int ky = tap / 3, kx = tap - ky * 3;
      const _Float16* bp = wpk + (size_t)(tap * 2 + cc) * 16 * 32;
      v16h Bf = *(const v16h*)(bp + (lidx * 32 + lhalf * 16));
      const _Float16* ap = &smem[(ky * 66 + (wave * 16 + lidx + kx)) * 32 + lhalf * 16];
      union { v16h v; v8h h[2]; } A;
      A.h[0] = *(const v8h*)ap;
      A.h[1] = *(const v8h*)(ap + 8);
      acc = __builtin_amdgcn_wmma_f32_16x16x32_f16(false, A.v, false, Bf,
                                                   (short)0, acc, false, false);
    }
  }
  if (lidx < ccnt) {
    float bv = bias[lidx];
#pragma unroll
    for (int r = 0; r < 8; ++r) {
      int gx = x0 + wave * 16 + r + lhalf * 8;
      if (gx < WW)
        out[(((size_t)b * 12 + choff + lidx) * HH + y) * WW + gx] = acc[r] + bv;
    }
  }
}

// ---------------- target assignment ----------------
__device__ __forceinline__ float gauss_radius(float h, float w, float ov) {
  float b1 = h + w, c1 = w * h * (1.f - ov) / (1.f + ov);
  float r1 = (b1 + sqrtf(fmaxf(b1 * b1 - 4.f * c1, 0.f))) * 0.5f;
  float b2 = 2.f * (h + w), c2 = (1.f - ov) * w * h;
  float r2 = (b2 + sqrtf(fmaxf(b2 * b2 - 16.f * c2, 0.f))) * 0.5f;
  float a3 = 4.f * ov, b3 = -2.f * ov * (h + w), c3 = (ov - 1.f) * w * h;
  float r3 = (b3 + sqrtf(fmaxf(b3 * b3 - 4.f * a3 * c3, 0.f))) * 0.5f;
  return fminf(fminf(r1, r2), r3);
}

__global__ __launch_bounds__(256) void box_kernel(const float* __restrict__ gt,
                                                  float* __restrict__ tb,
                                                  float* __restrict__ inds,
                                                  float* __restrict__ mask,
                                                  float* __restrict__ bp) {
  int i = blockIdx.x * 256 + threadIdx.x;
  if (i >= NB * NMAX) return;
  int b = i / NMAX, m = i - b * NMAX;
  if (m >= NM) {
#pragma unroll
    for (int j = 0; j < 8; ++j) tb[i * 8 + j] = 0.f;
    inds[i] = 0.f;
    mask[i] = 0.f;
    return;
  }
  const float* g = &gt[(b * NM + m) * 8];
  float x = g[0], y = g[1], z = g[2], dx = g[3], dy = g[4], dz = g[5], hd = g[6];
  float clsf = g[7];
  float cx = fminf(fmaxf((x + 75.2f) * 10.f * 0.125f, 0.f), WW - 0.5f);
  float cy = fminf(fmaxf((y + 75.2f) * 10.f * 0.125f, 0.f), HH - 0.5f);
  float cxi = floorf(cx), cyi = floorf(cy);
  float dxp = dx * 1.25f;
  float dyp = dy * 1.25f;
  float r = fmaxf(floorf(gauss_radius(dxp, dyp, 0.1f)), 2.0f);
  float sigma = (2.f * r + 1.f) * (1.f / 6.f);
  bool valid = (dx > 0.f) && (dy > 0.f);
  float vf = valid ? 1.f : 0.f;
  tb[i * 8 + 0] = (cx - cxi) * vf;
  tb[i * 8 + 1] = (cy - cyi) * vf;
  tb[i * 8 + 2] = z * vf;
  tb[i * 8 + 3] = logf(dx) * vf;
  tb[i * 8 + 4] = logf(dy) * vf;
  tb[i * 8 + 5] = logf(dz) * vf;
  tb[i * 8 + 6] = cosf(hd) * vf;
  tb[i * 8 + 7] = sinf(hd) * vf;
  inds[i] = valid ? (float)((int)cyi * WW + (int)cxi) : 0.f;
  mask[i] = vf;
  float* p = &bp[(b * NM + m) * 6];
  p[0] = cxi; p[1] = cyi; p[2] = r;
  p[3] = 1.f / (2.f * sigma * sigma);
  p[4] = clsf; p[5] = vf;
}

__global__ __launch_bounds__(256) void heatmap_kernel(const float* __restrict__ bp,
                                                      float* __restrict__ hm) {
  __shared__ float sb[NM * 6];
  int b = blockIdx.z, c = blockIdx.y;
  for (int i = threadIdx.x; i < NM * 6; i += 256) sb[i] = bp[b * NM * 6 + i];
  __syncthreads();
  int pix = blockIdx.x * 256 + threadIdx.x;
  if (pix >= HH * WW) return;
  int y = pix / WW, x = pix - y * WW;
  float best = 0.f;
  for (int m = 0; m < NM; ++m) {
    const float* p = &sb[m * 6];
    if (p[5] == 0.f) continue;
    if ((int)p[4] != c + 1) continue;
    float ox = (float)x - p[0];
    float oy = (float)y - p[1];
    float r = p[2];
    if (fabsf(ox) > r || fabsf(oy) > r) continue;
    float gsn = __expf(-(ox * ox + oy * oy) * p[3]);
    best = fmaxf(best, gsn);
  }
  hm[((b * NCLS + c) * HH + y) * WW + x] = best;
}

// ---------------- launch ----------------
extern "C" void kernel_launch(void* const* d_in, const int* in_sizes, int n_in,
                              void* d_out, int out_size, void* d_ws, size_t ws_size,
                              hipStream_t stream) {
  const float* spatial = (const float*)d_in[0];
  const float* gt      = (const float*)d_in[1];
  const float* w_sh    = (const float*)d_in[2];
  const float* bn_sh   = (const float*)d_in[3];
  const float* w1      = (const float*)d_in[4];
  const float* bn_h    = (const float*)d_in[5];
  const float* w2      = (const float*)d_in[6];
  const float* b2      = (const float*)d_in[7];
  float* out = (float*)d_out;

  const size_t HM_OFF   = (size_t)NB * 12 * HH * WW;
  const size_t TB_OFF   = HM_OFF + (size_t)NB * NCLS * HH * WW;
  const size_t IND_OFF  = TB_OFF + (size_t)NB * NMAX * 8;
  const size_t MSK_OFF  = IND_OFF + (size_t)NB * NMAX;

  char* ws = (char*)d_ws;
  _Float16* pws = (_Float16*)(ws);                       // 294912 h
  _Float16* pw1 = (_Float16*)(ws + 589824);              // 221184 h
  _Float16* pw2 = (_Float16*)(ws + 1032192);             // 55296 h
  float*    bparam = (float*)(ws + 1142784);             // 4800 f32
  _Float16* sact = (_Float16*)(ws + 1161984);            // 9048064 h
  _Float16* h1   = (_Float16*)(ws + 1161984 + 18096128); // 9048064 h

  pack_shared<<<(9 * 16 * 64 * 32 + 255) / 256, 256, 0, stream>>>(w_sh, pws);
  pack_head1<<<(6 * 9 * 2 * 64 * 32 + 255) / 256, 256, 0, stream>>>(w1, pw1);
  pack_head2<<<(6 * 9 * 2 * 16 * 32 + 255) / 256, 256, 0, stream>>>(w2, pw2);

  dim3 cgrid(3, HH, NB);
  conv_shared<<<cgrid, 128, 0, stream>>>(spatial, pws, bn_sh, sact);

  static const int choff[6] = {0, 2, 3, 6, 8, 9};
  static const int ccnt[6]  = {2, 1, 3, 2, 1, 3};
  for (int head = 0; head < 6; ++head) {
    conv_head1<<<cgrid, 128, 0, stream>>>(
        sact, pw1 + (size_t)head * 9 * 2 * 64 * 32, bn_h + head * 4 * 64, h1);
    conv_head2<<<cgrid, 128, 0, stream>>>(
        h1, pw2 + (size_t)head * 9 * 2 * 16 * 32, b2 + head * 3,
        out, choff[head], ccnt[head]);
  }

  box_kernel<<<(NB * NMAX + 255) / 256, 256, 0, stream>>>(
      gt, out + TB_OFF, out + IND_OFF, out + MSK_OFF, bparam);
  dim3 hmgrid((HH * WW + 255) / 256, NCLS, NB);
  heatmap_kernel<<<hmgrid, 256, 0, stream>>>(bparam, out + HM_OFF);
}
